// ContextSeq2Mat_10539849744799
// MI455X (gfx1250) — matrix operand order repaired and checked by reading, not compiled, not run on hardware
//
#include <hip/hip_runtime.h>
#include <hip/hip_bf16.h>
#include <math.h>

typedef __attribute__((ext_vector_type(16))) _Float16 v16h;
typedef __attribute__((ext_vector_type(8)))  _Float16 v8h;
typedef __attribute__((ext_vector_type(8)))  float    v8f;
typedef __attribute__((ext_vector_type(4)))  unsigned int v4u;

#define BB 2
#define LL 512
#define HH 128
#define NLVL 10   // 2^9 = 512 max span

// packed f16 max: raw v_pk_max_num_f16, no NaN-canonicalize (inputs finite)
static __device__ __forceinline__ v8h hmax8(v8h a, v8h b) {
  v4u ua = __builtin_bit_cast(v4u, a);
  v4u ub = __builtin_bit_cast(v4u, b);
  v4u ur;
#pragma unroll
  for (int q = 0; q < 4; ++q) {
    unsigned int r;
    asm("v_pk_max_num_f16 %0, %1, %2" : "=v"(r) : "v"(ua[q]), "v"(ub[q]));
    ur[q] = r;
  }
  return __builtin_bit_cast(v8h, ur);
}

// ---- tiny prologue: per-row fp32 projections a = x@Wx^T, bvec = y@Wy^T ----
__global__ void proj_kernel(const float* __restrict__ x, const float* __restrict__ y,
                            const float* __restrict__ W, float* __restrict__ aout,
                            float* __restrict__ bout) {
  int idx = blockIdx.x * blockDim.x + threadIdx.x;   // over B*L*H
  int g  = idx % HH;
  int bl = idx / HH;                                  // b*L + l
  const float* xr = x + (size_t)bl * HH;
  const float* yr = y + (size_t)bl * HH;
  const float* wx = W + (size_t)g * (3 * HH);
  const float* wy = wx + HH;
  float sa = 0.f, sb = 0.f;
#pragma unroll 8
  for (int h = 0; h < HH; ++h) {
    sa = fmaf(xr[h], wx[h], sa);
    sb = fmaf(yr[h], wy[h], sb);
  }
  aout[idx] = sa;
  bout[idx] = sb;
}

// ---- sparse table level 0: x -> f16 (max commutes with monotone rounding) ----
__global__ void cvt_kernel(const float* __restrict__ x, _Float16* __restrict__ P) {
  int idx = blockIdx.x * blockDim.x + threadIdx.x;   // B*L*H
  P[idx] = (_Float16)x[idx];
}

// ---- sparse table level build: P[l][k] = max(P[l-1][k], P[l-1][k+2^(l-1)]) ----
__global__ void spmax_kernel(_Float16* __restrict__ P, int lvl) {
  int idx = blockIdx.x * blockDim.x + threadIdx.x;   // B*L*H
  int h = idx % HH;
  int k = (idx / HH) % LL;
  int b = idx / (HH * LL);
  int kk = k + (1 << (lvl - 1));
  if (kk > LL - 1) kk = LL - 1;                      // clamp (entries only read when valid)
  const _Float16* prev = P + (size_t)(lvl - 1) * BB * LL * HH;
  _Float16 va = prev[((size_t)b * LL + k)  * HH + h];
  _Float16 vb = prev[((size_t)b * LL + kk) * HH + h];
  P[(size_t)lvl * BB * LL * HH + idx] = (va > vb) ? va : vb;
}

// ---- main fused kernel: span-max ctx -> f16 WMMA GEMM -> +proj+bias -> gelu ----
__global__ void __launch_bounds__(256)
ctx_gemm_kernel(const _Float16* __restrict__ Ph, const float* __restrict__ W,
                const float* __restrict__ aproj, const float* __restrict__ bproj,
                const float* __restrict__ bias, float* __restrict__ out) {
  // Wc^T in f16, pre-swizzled into WMMA B-fragment layout:
  // fragment f = ntile*4+kc, lane L, half t  ->  k = kc*32 + 8*(L/16) + (t&7) + 16*(t>>3), n = ntile*16 + L%16
  __shared__ _Float16 smB[8 * 4 * 32 * 16];          // 16384 halfs = 32 KB

  const int tid  = threadIdx.x;
  const int wave = tid >> 5;
  const int lane = tid & 31;

  const int bx = blockIdx.x;
  const int jt = bx & 31;                            // 32 j-tiles of 16
  const int it = (bx >> 5) & 63;                     // 64 i-tiles of 8 (one i per wave)
  const int bb = bx >> 11;                           // batch

  for (int e = tid; e < 8 * 4 * 32 * 16; e += 256) {
    int t  = e & 15;
    int Lz = (e >> 4) & 31;
    int f  = e >> 9;                                 // ntile*4 + kc
    int kc = f & 3;
    int nt = f >> 2;
    int k  = kc * 32 + ((Lz >> 4) << 3) + (t & 7) + ((t >> 3) << 4);
    int n  = nt * 16 + (Lz & 15);
    smB[e] = (_Float16)W[(size_t)n * (3 * HH) + 2 * HH + k];
  }
  __syncthreads();

  const int i  = it * 8 + wave;
  const int j0 = jt * 16;
  const int m  = lane & 15;                          // A-matrix row owned by this lane
  const int hb = (lane >> 4) << 3;                   // K sub-base: 0 or 8
  const int j  = j0 + m;

  // span [lo, hi] lookup rows in sparse table
  int d = i - j; if (d < 0) d = -d;
  int len = d + 1;
  int lvl = 31 - __clz(len);
  int lo  = (i < j) ? i : j;
  int r2  = lo + len - (1 << lvl);                   // hi - 2^lvl + 1

  const _Float16* rowA = Ph + ((size_t)lvl * BB * LL + (size_t)bb * LL + lo) * HH;
  const _Float16* rowB = Ph + ((size_t)lvl * BB * LL + (size_t)bb * LL + r2) * HH;

  // A fragments (16x32 f16 each): halves t<8 -> h = kc*32+hb+t ; t>=8 -> h = kc*32+16+hb+(t-8)
  v16h afrag[4];
#pragma unroll
  for (int kc = 0; kc < 4; ++kc) {
    int o0 = kc * 32 + hb;
    v8h a0 = *(const v8h*)(rowA + o0);
    v8h a1 = *(const v8h*)(rowA + o0 + 16);
    v8h b0 = *(const v8h*)(rowB + o0);
    v8h b1 = *(const v8h*)(rowB + o0 + 16);
    v8h m0 = hmax8(a0, b0);
    v8h m1 = hmax8(a1, b1);
#pragma unroll
    for (int q = 0; q < 8; ++q) { afrag[kc][q] = m0[q]; afrag[kc][q + 8] = m1[q]; }
  }

  v8f acc[8];
#pragma unroll
  for (int nt = 0; nt < 8; ++nt)
#pragma unroll
    for (int q = 0; q < 8; ++q) acc[nt][q] = 0.f;

#pragma unroll
  for (int nt = 0; nt < 8; ++nt) {
#pragma unroll
    for (int kc = 0; kc < 4; ++kc) {
      const v16h bfrag = *(const v16h*)(&smB[(((nt << 2) | kc) * 32 + lane) << 4]);
      acc[nt] = __builtin_amdgcn_wmma_f32_16x16x32_f16(
          false, afrag[kc], false, bfrag, (short)0, acc[nt], false, false);
    }
  }

  // epilogue: C/D layout -> VGPR r holds M = r + 8*(lane/16), N = lane%16
  const int rowoff = (lane >> 4) << 3;
#pragma unroll
  for (int nt = 0; nt < 8; ++nt) {
    const int g   = nt * 16 + (lane & 15);
    const float av = aproj[((size_t)bb * LL + i) * HH + g];
    const float bs = bias[g];
#pragma unroll
    for (int rr = 0; rr < 8; ++rr) {
      const int jm = j0 + rr + rowoff;
      const float bv = bproj[((size_t)bb * LL + jm) * HH + g];
      float tv = acc[nt][rr] + av + bv + bs;
      float gl = 0.5f * tv * (1.0f + erff(tv * 0.70710678118654752f));  // exact gelu
      out[(((size_t)bb * LL + i) * LL + jm) * HH + g] = gl;
    }
  }
}

extern "C" void kernel_launch(void* const* d_in, const int* in_sizes, int n_in,
                              void* d_out, int out_size, void* d_ws, size_t ws_size,
                              hipStream_t stream) {
  const float* x    = (const float*)d_in[0];
  const float* y    = (const float*)d_in[1];
  const float* W    = (const float*)d_in[2];
  const float* bias = (const float*)d_in[3];
  float* out = (float*)d_out;

  char* ws = (char*)d_ws;
  _Float16* Ph   = (_Float16*)ws;                            // 10*2*512*128*2 = 2,621,440 B
  float*    aprj = (float*)(ws + 2621440);                   //   524,288 B
  float*    bprj = (float*)(ws + 2621440 + 524288);          //   524,288 B

  const int n = BB * LL * HH;                                // 131072
  proj_kernel<<<n / 256, 256, 0, stream>>>(x, y, W, aprj, bprj);
  cvt_kernel<<<n / 256, 256, 0, stream>>>(x, Ph);
  for (int l = 1; l < NLVL; ++l)
    spmax_kernel<<<n / 256, 256, 0, stream>>>(Ph, l);

  ctx_gemm_kernel<<<BB * 64 * 32, 256, 0, stream>>>(Ph, W, aprj, bprj, bias, out);
}
